// CustomTransformerEncoderLayer_71442486002223
// MI455X (gfx1250) — matrix operand order repaired and checked by reading, not compiled, not run on hardware
//
#include <hip/hip_runtime.h>

#if defined(__has_builtin)
#if __has_builtin(__builtin_amdgcn_tensor_load_to_lds) && \
    __has_builtin(__builtin_amdgcn_s_wait_tensorcnt)
#define HAVE_TDM 1
#endif
#endif
#ifndef HAVE_TDM
#define HAVE_TDM 0
#endif

// ---------------------------------------------------------------------------
// Types
// ---------------------------------------------------------------------------
typedef __bf16 bf16;
typedef __attribute__((ext_vector_type(16))) __bf16 v16bf;
typedef __attribute__((ext_vector_type(8)))  __bf16 v8bf;
typedef __attribute__((ext_vector_type(8)))  float  v8f;
typedef __attribute__((ext_vector_type(4)))  unsigned int u32x4;
typedef __attribute__((ext_vector_type(8)))  int i32x8;
typedef __attribute__((ext_vector_type(4)))  int i32x4;

static __device__ __forceinline__ bf16 f2bf(float f) {
  union { float f; unsigned u; } v; v.f = f;
  unsigned r = v.u + 0x7FFFu + ((v.u >> 16) & 1u);   // round-to-nearest-even
  unsigned short h = (unsigned short)(r >> 16);
  return __builtin_bit_cast(bf16, h);
}

static __device__ __forceinline__ v8f vzero8() {
  v8f z;
#pragma unroll
  for (int i = 0; i < 8; ++i) z[i] = 0.0f;
  return z;
}

// A-fragment (16x32 bf16): per lane two contiguous 8-half (16B) chunks.
static __device__ __forceinline__ v16bf ld_frag_a(const bf16* p0, const bf16* p1) {
  v8bf lo = *(const v8bf*)p0;
  v8bf hi = *(const v8bf*)p1;
  return __builtin_shufflevector(lo, hi, 0,1,2,3,4,5,6,7,8,9,10,11,12,13,14,15);
}
// B-fragment (32x16 bf16): per lane one contiguous 16-half (32B) chunk.
static __device__ __forceinline__ v16bf ld_frag_b(const bf16* p) {
  v8bf lo = *(const v8bf*)p;
  v8bf hi = *(const v8bf*)(p + 8);
  return __builtin_shufflevector(lo, hi, 0,1,2,3,4,5,6,7,8,9,10,11,12,13,14,15);
}

#if HAVE_TDM
// LDS byte offset of a __shared__ object: flat address low 32 bits are the
// LDS logical address (ISA 10.2 aperture mapping).
static __device__ __forceinline__ unsigned lds_off(const void* p) {
  return (unsigned)(unsigned long long)p;
}

// 2D bf16 tile DMA: tile_h rows x tile_w elems from row-major [*, row_stride]
// matrix into LDS at lds_byte_off, with LDS row padding (pad codes per D#).
static __device__ __forceinline__ void tdm_load_2d_bf16(
    unsigned lds_byte_off, const bf16* gsrc,
    unsigned tensor_w, unsigned tensor_h,
    unsigned tile_w, unsigned tile_h, unsigned row_stride_elems,
    unsigned pad_interval_code, unsigned pad_amount_code)
{
  const unsigned long long ga = (unsigned long long)(const void*)gsrc;
  u32x4 g0;
  g0[0] = 1u;                                            // count=1 (valid D#)
  g0[1] = lds_byte_off;                                  // lds_addr
  g0[2] = (unsigned)ga;                                  // global_addr[31:0]
  g0[3] = (unsigned)((ga >> 32) & 0x01FFFFFFu)           // global_addr[56:32]
        | 0x80000000u;                                   // type=2 ("image")
  i32x8 g1;
  unsigned w0 = (1u << 16)                               // data_size=1 -> 2B
              | (1u << 20)                               // pad_enable
              | (pad_interval_code << 22)
              | (pad_amount_code << 25);
  g1[0] = (int)w0;
  g1[1] = (int)((tensor_w & 0xFFFFu) << 16);             // tensor_dim0[15:0]
  g1[2] = (int)(((tensor_w >> 16) & 0xFFFFu)             // tensor_dim0[31:16]
              | ((tensor_h & 0xFFFFu) << 16));           // tensor_dim1[15:0]
  g1[3] = (int)(((tensor_h >> 16) & 0xFFFFu)             // tensor_dim1[31:16]
              | ((tile_w & 0xFFFFu) << 16));             // tile_dim0
  g1[4] = (int)(tile_h & 0xFFFFu);                       // tile_dim1 (tile_dim2=0)
  g1[5] = (int)row_stride_elems;                         // tensor_dim0_stride[31:0]
  g1[6] = 0;                                             // stride0[47:32], stride1 lo
  g1[7] = 0;
  i32x4 gz; gz[0] = gz[1] = gz[2] = gz[3] = 0;           // 2D: groups 2/3 unused
  i32x8 gz8;
#pragma unroll
  for (int i = 0; i < 8; ++i) gz8[i] = 0;
  __builtin_amdgcn_tensor_load_to_lds(g0, g1, gz, gz, gz8, 0);
}
#endif // HAVE_TDM

// ---------------------------------------------------------------------------
// fp32 -> bf16 conversion
// ---------------------------------------------------------------------------
__global__ __launch_bounds__(256)
void cvt_f32_bf16(const float* __restrict__ in, bf16* __restrict__ out, int n) {
  int i = blockIdx.x * 256 + threadIdx.x;
  if (i < n) out[i] = f2bf(in[i]);
}

// ---------------------------------------------------------------------------
// GEMM: C[M,N] = A[M,K](bf16) @ W[N,K](bf16)^T + bias (+ residual fp32) (+ReLU)
// Block tile 128x128, 256 threads = 8 waves, wave tile 64x32 (4x2 WMMAs).
// Tile staging: TDM (TENSOR_LOAD_TO_LDS, double-buffered, padded rows) when
// available, else synchronous cooperative loads.
// ---------------------------------------------------------------------------
template<bool RELU>
__global__ __launch_bounds__(256)
void gemm_bf16_nt(const bf16* __restrict__ A, const bf16* __restrict__ W,
                  const float* __restrict__ bias, const float* __restrict__ resid,
                  float* __restrict__ Cf, bf16* __restrict__ Cbf,
                  int M, int N, int K)
{
  constexpr int LDR = 40;            // LDS row stride (halves): 64B data + 16B pad
#if HAVE_TDM
  __shared__ bf16 sA[2][128 * LDR];
  __shared__ bf16 sB[2][128 * LDR];
#else
  __shared__ bf16 sA[1][128 * LDR];
  __shared__ bf16 sB[1][128 * LDR];
#endif

  const int tid  = threadIdx.x;
  const int lane = tid & 31;
  const int wv   = tid >> 5;         // 0..7
  const int wm   = wv & 1;           // wave row (2)
  const int wn   = wv >> 1;          // wave col (4)
  const int l15  = lane & 15;
  const int lh   = lane >> 4;        // 0/1

  const int baseM = blockIdx.y * 128;
  const int baseN = blockIdx.x * 128;

  v8f acc[4][2];
#pragma unroll
  for (int i = 0; i < 4; ++i)
#pragma unroll
    for (int j = 0; j < 2; ++j) acc[i][j] = vzero8();

#if HAVE_TDM
  // Prologue: DMA first K-tiles (pad: 64B interval (code 3), 4 DWORDs (code 3)).
  if (wv == 0) {
    tdm_load_2d_bf16(lds_off(&sA[0][0]), A + (size_t)baseM * K,
                     (unsigned)K, 1u << 20, 32, 128, (unsigned)K, 3, 3);
    tdm_load_2d_bf16(lds_off(&sB[0][0]), W + (size_t)baseN * K,
                     (unsigned)K, 1u << 20, 32, 128, (unsigned)K, 3, 3);
    __builtin_amdgcn_s_wait_tensorcnt(0);
  }
  __syncthreads();
#endif

  int cur = 0;
  for (int k0 = 0; k0 < K; k0 += 32) {
#if HAVE_TDM
    // Kick DMA for the next K-tiles into the other buffer, then compute.
    if (k0 + 32 < K && wv == 0) {
      tdm_load_2d_bf16(lds_off(&sA[cur ^ 1][0]), A + (size_t)baseM * K + k0 + 32,
                       (unsigned)K, 1u << 20, 32, 128, (unsigned)K, 3, 3);
      tdm_load_2d_bf16(lds_off(&sB[cur ^ 1][0]), W + (size_t)baseN * K + k0 + 32,
                       (unsigned)K, 1u << 20, 32, 128, (unsigned)K, 3, 3);
    }
#else
    // -------- synchronous staging of 128x32 tiles (16B chunks) --------
#pragma unroll
    for (int i = 0; i < 2; ++i) {
      int c  = tid + 256 * i;        // 0..511
      int r  = c >> 2;               // 0..127
      int cc = (c & 3) * 8;          // half offset 0/8/16/24
      *(uint4*)(&sA[0][0] + r * LDR + cc) =
          *(const uint4*)(A + (size_t)(baseM + r) * K + k0 + cc);
      *(uint4*)(&sB[0][0] + r * LDR + cc) =
          *(const uint4*)(W + (size_t)(baseN + r) * K + k0 + cc);
    }
    if (k0 + 32 < K) {
      __builtin_prefetch(A + (size_t)(baseM + (tid >> 1)) * K + k0 + 32, 0, 1);
      __builtin_prefetch(W + (size_t)(baseN + (tid >> 1)) * K + k0 + 32, 0, 1);
    }
    __syncthreads();
#endif

    // -------- fragments + 8 WMMAs --------
    v16bf af[4], bfg[2];
#pragma unroll
    for (int i = 0; i < 4; ++i) {
      const bf16* base = &sA[cur][0] + (size_t)(wm * 64 + i * 16 + l15) * LDR;
      af[i] = ld_frag_a(base + lh * 16, base + lh * 16 + 8);
    }
#pragma unroll
    for (int j = 0; j < 2; ++j) {
      const bf16* base = &sB[cur][0] + (size_t)(wn * 32 + j * 16 + l15) * LDR;
      bfg[j] = ld_frag_b(base + lh * 16);
    }
#pragma unroll
    for (int i = 0; i < 4; ++i)
#pragma unroll
      for (int j = 0; j < 2; ++j)
        acc[i][j] = __builtin_amdgcn_wmma_f32_16x16x32_bf16(
            false, af[i], false, bfg[j], (short)0, acc[i][j], false, false);

#if HAVE_TDM
    if (wv == 0) __builtin_amdgcn_s_wait_tensorcnt(0);
    __syncthreads();
    cur ^= 1;
#else
    __syncthreads();
#endif
  }

  // -------- epilogue: C layout lane l: row = r + 8*(l>=16), col = l&15 ------
  const int mo = lh * 8;
#pragma unroll
  for (int i = 0; i < 4; ++i) {
#pragma unroll
    for (int j = 0; j < 2; ++j) {
      const int n  = baseN + wn * 32 + j * 16 + l15;
      const float bv = bias ? bias[n] : 0.0f;
#pragma unroll
      for (int r = 0; r < 8; ++r) {
        const int m = baseM + wm * 64 + i * 16 + mo + r;
        float v = acc[i][j][r] + bv;
        const size_t idx = (size_t)m * N + n;
        if (resid) v += resid[idx];
        if (RELU)  v = v > 0.0f ? v : 0.0f;
        if (Cf)    Cf[idx]  = v;
        if (Cbf)   Cbf[idx] = f2bf(v);
      }
    }
  }
}

// ---------------------------------------------------------------------------
// RoPE + split fp32 QKV [M,3D] -> head-major bf16 q/k/v [B,H,S,Dh]
// ---------------------------------------------------------------------------
__global__ __launch_bounds__(256)
void rope_split_kernel(const float* __restrict__ qkv,
                       bf16* __restrict__ qh, bf16* __restrict__ kh,
                       bf16* __restrict__ vh)
{
  const int t = blockIdx.x * 256 + threadIdx.x;
  const int i = t & 31;              // rope pair 0..31
  const int h = (t >> 5) & 15;      // head
  const int s = (t >> 9) & 2047;    // seq pos
  const int b = t >> 20;            // batch

  const float inv_freq = __expf(-(float)(2 * i) * (9.210340371976184f / 64.0f));
  const float ang = (float)s * inv_freq;
  float sn, cs;
  __sincosf(ang, &sn, &cs);

  const size_t base = (size_t)(b * 2048 + s) * 3072;
  const int off = h * 64 + 2 * i;
  const float qe = qkv[base + off],        qo = qkv[base + off + 1];
  const float ke = qkv[base + 1024 + off], ko = qkv[base + 1024 + off + 1];
  const float ve = qkv[base + 2048 + off], vo = qkv[base + 2048 + off + 1];

  const size_t dst = (size_t)((b * 16 + h) * 2048 + s) * 64 + 2 * i;
  qh[dst]     = f2bf(qe * cs - qo * sn);
  qh[dst + 1] = f2bf(qe * sn + qo * cs);
  kh[dst]     = f2bf(ke * cs - ko * sn);
  kh[dst + 1] = f2bf(ke * sn + ko * cs);
  vh[dst]     = f2bf(ve);
  vh[dst + 1] = f2bf(vo);
}

// ---------------------------------------------------------------------------
// Flash attention: one block per (q-tile 64, head, batch); 4 waves,
// each wave owns 16 q rows.  K tiles of 64, online softmax in fp32.
// ---------------------------------------------------------------------------
__global__ __launch_bounds__(128)
void flash_attn_kernel(const bf16* __restrict__ q, const bf16* __restrict__ k,
                       const bf16* __restrict__ v, bf16* __restrict__ attn)
{
  constexpr int LR = 72;             // LDS row stride (halves): 144B, 16B aligned
  __shared__ bf16 sK [64 * LR];      // K tile  [sk][dh]
  __shared__ bf16 sVt[64 * LR];      // V^T     [dh][sk]
  __shared__ bf16 sP [64 * LR];      // P stage, 16 rows per wave

  const int tid  = threadIdx.x;
  const int lane = tid & 31;
  const int wv   = tid >> 5;         // 0..3
  const int l15  = lane & 15;
  const int lh   = lane >> 4;

  const int q0 = blockIdx.x * 64;
  const int hh = blockIdx.y;
  const int bb = blockIdx.z;
  const size_t headoff = (size_t)(bb * 16 + hh) * 2048 * 64;
  const bf16* qg = q + headoff;
  const bf16* kg = k + headoff;
  const bf16* vg = v + headoff;

  // preload Q fragments for this wave's 16 rows (Dh=64 -> 2 K-chunks)
  v16bf qf[2];
  {
    const bf16* qrow = qg + (size_t)(q0 + wv * 16 + l15) * 64;
#pragma unroll
    for (int kc = 0; kc < 2; ++kc)
      qf[kc] = ld_frag_a(qrow + kc * 32 + lh * 16, qrow + kc * 32 + lh * 16 + 8);
  }

  float mrow[8], lrow[8];
  v8f   oacc[4];
#pragma unroll
  for (int r = 0; r < 8; ++r) { mrow[r] = -1e30f; lrow[r] = 0.0f; }
#pragma unroll
  for (int j = 0; j < 4; ++j) oacc[j] = vzero8();

  const float scale = 0.125f;        // 64^-0.5

  for (int kt = 0; kt < 32; ++kt) {
    // ---- stage K tile and transposed V tile ----
#pragma unroll
    for (int i = 0; i < 2; ++i) {
      int c   = tid + 128 * i;       // 0..255
      int sk  = c >> 2;              // 0..63
      int off = (c & 3) * 8;
      *(uint4*)(sK + sk * LR + off) =
          *(const uint4*)(kg + (size_t)(kt * 64 + sk) * 64 + off);
      v8bf vv = *(const v8bf*)(vg + (size_t)(kt * 64 + sk) * 64 + off);
#pragma unroll
      for (int u = 0; u < 8; ++u) sVt[(off + u) * LR + sk] = vv[u];
    }
    __syncthreads();

    // ---- scores: 16x64 per wave via 8 WMMAs ----
    v8f sc[4];
#pragma unroll
    for (int j = 0; j < 4; ++j) sc[j] = vzero8();
#pragma unroll
    for (int kc = 0; kc < 2; ++kc) {
#pragma unroll
      for (int j = 0; j < 4; ++j) {
        v16bf bfrag = ld_frag_b(sK + (size_t)(j * 16 + l15) * LR + kc * 32 + lh * 16);
        sc[j] = __builtin_amdgcn_wmma_f32_16x16x32_bf16(
            false, qf[kc], false, bfrag, (short)0, sc[j], false, false);
      }
    }
#pragma unroll
    for (int j = 0; j < 4; ++j)
#pragma unroll
      for (int r = 0; r < 8; ++r) sc[j][r] *= scale;

    // ---- online softmax: row max / rescale (half-wave shuffles) ----
    float alpha[8];
#pragma unroll
    for (int r = 0; r < 8; ++r) {
      float vm = fmaxf(fmaxf(sc[0][r], sc[1][r]), fmaxf(sc[2][r], sc[3][r]));
      vm = fmaxf(vm, __shfl_xor(vm, 1, 32));
      vm = fmaxf(vm, __shfl_xor(vm, 2, 32));
      vm = fmaxf(vm, __shfl_xor(vm, 4, 32));
      vm = fmaxf(vm, __shfl_xor(vm, 8, 32));
      const float mn = fmaxf(mrow[r], vm);
      alpha[r] = __expf(mrow[r] - mn);
      mrow[r]  = mn;
    }
    float rs[8];
#pragma unroll
    for (int r = 0; r < 8; ++r) rs[r] = 0.0f;
#pragma unroll
    for (int j = 0; j < 4; ++j)
#pragma unroll
      for (int r = 0; r < 8; ++r) {
        const float p = __expf(sc[j][r] - mrow[r]);
        rs[r] += p;
        sP[(size_t)(wv * 16 + lh * 8 + r) * LR + j * 16 + l15] = f2bf(p);
      }
#pragma unroll
    for (int r = 0; r < 8; ++r) {
      float t = rs[r];
      t += __shfl_xor(t, 1, 32);
      t += __shfl_xor(t, 2, 32);
      t += __shfl_xor(t, 4, 32);
      t += __shfl_xor(t, 8, 32);
      lrow[r] = lrow[r] * alpha[r] + t;
#pragma unroll
      for (int j = 0; j < 4; ++j) oacc[j][r] *= alpha[r];
    }

    // ---- O += P @ V  (contraction over sk, two 32-chunks) ----
#pragma unroll
    for (int kc = 0; kc < 2; ++kc) {
      const bf16* pr = sP + (size_t)(wv * 16 + l15) * LR + kc * 32;
      v16bf pa = ld_frag_a(pr + lh * 16, pr + lh * 16 + 8);
#pragma unroll
      for (int j = 0; j < 4; ++j) {
        v16bf bfrag = ld_frag_b(sVt + (size_t)(j * 16 + l15) * LR + kc * 32 + lh * 16);
        oacc[j] = __builtin_amdgcn_wmma_f32_16x16x32_bf16(
            false, pa, false, bfrag, (short)0, oacc[j], false, false);
      }
    }
    __syncthreads();
  }

  // ---- normalize + write attn output, token-major [B,S,H*Dh] bf16 ----
#pragma unroll
  for (int j = 0; j < 4; ++j)
#pragma unroll
    for (int r = 0; r < 8; ++r) {
      const float val = oacc[j][r] / lrow[r];
      const int row = q0 + wv * 16 + lh * 8 + r;
      const int dh  = j * 16 + l15;
      attn[(size_t)(bb * 2048 + row) * 1024 + hh * 64 + dh] = f2bf(val);
    }
}

// ---------------------------------------------------------------------------
// LayerNorm over D=1024; one block per row, 256 threads x 4 elements.
// ---------------------------------------------------------------------------
__global__ __launch_bounds__(256)
void layernorm_kernel(const float* __restrict__ in, const float* __restrict__ g,
                      const float* __restrict__ bta, float* __restrict__ outf,
                      bf16* __restrict__ outbf)
{
  __shared__ float red[16];
  __shared__ float stats[2];
  const int row = blockIdx.x;
  const int tid = threadIdx.x;

  const float4 v4 = *(const float4*)(in + (size_t)row * 1024 + tid * 4);
  float s  = v4.x + v4.y + v4.z + v4.w;
  float ss = v4.x * v4.x + v4.y * v4.y + v4.z * v4.z + v4.w * v4.w;
#pragma unroll
  for (int m = 1; m <= 16; m <<= 1) {
    s  += __shfl_xor(s,  m, 32);
    ss += __shfl_xor(ss, m, 32);
  }
  if ((tid & 31) == 0) { red[tid >> 5] = s; red[8 + (tid >> 5)] = ss; }
  __syncthreads();
  if (tid == 0) {
    float ts = 0.0f, tss = 0.0f;
#pragma unroll
    for (int w = 0; w < 8; ++w) { ts += red[w]; tss += red[8 + w]; }
    const float mu  = ts * (1.0f / 1024.0f);
    const float var = tss * (1.0f / 1024.0f) - mu * mu;
    stats[0] = mu;
    stats[1] = rsqrtf(var + 1e-5f);
  }
  __syncthreads();
  const float mu = stats[0], rsg = stats[1];
  const float xv[4] = { v4.x, v4.y, v4.z, v4.w };
#pragma unroll
  for (int e = 0; e < 4; ++e) {
    const int col = tid * 4 + e;
    const float y = (xv[e] - mu) * rsg * g[col] + bta[col];
    const size_t idx = (size_t)row * 1024 + col;
    if (outf)  outf[idx]  = y;
    if (outbf) outbf[idx] = f2bf(y);
  }
}

// ---------------------------------------------------------------------------
// Host launch
// ---------------------------------------------------------------------------
extern "C" void kernel_launch(void* const* d_in, const int* in_sizes, int n_in,
                              void* d_out, int out_size, void* d_ws, size_t ws_size,
                              hipStream_t stream)
{
  (void)in_sizes; (void)n_in; (void)out_size; (void)ws_size;
  const float* x    = (const float*)d_in[0];
  const float* wqkv = (const float*)d_in[1];
  const float* bqkv = (const float*)d_in[2];
  const float* wout = (const float*)d_in[3];
  const float* bout = (const float*)d_in[4];
  const float* w1   = (const float*)d_in[5];
  const float* b1   = (const float*)d_in[6];
  const float* w2   = (const float*)d_in[7];
  const float* b2   = (const float*)d_in[8];
  const float* ln1g = (const float*)d_in[9];
  const float* ln1b = (const float*)d_in[10];
  const float* ln2g = (const float*)d_in[11];
  const float* ln2b = (const float*)d_in[12];
  float* out = (float*)d_out;

  const int Mtok = 4096, D = 1024, F = 4096;

  char* p = (char*)d_ws;
  auto alloc = [&](size_t bytes) {
    char* r = p;
    p += (bytes + 255) & ~(size_t)255;
    return (void*)r;
  };
  bf16*  x_bf    = (bf16*) alloc((size_t)Mtok * D * 2);
  bf16*  wqkv_bf = (bf16*) alloc((size_t)3 * D * D * 2);
  bf16*  wout_bf = (bf16*) alloc((size_t)D * D * 2);
  bf16*  w1_bf   = (bf16*) alloc((size_t)F * D * 2);
  bf16*  w2_bf   = (bf16*) alloc((size_t)D * F * 2);
  float* qkv     = (float*)alloc((size_t)Mtok * 3 * D * 4);
  bf16*  q_bf    = (bf16*) alloc((size_t)Mtok * D * 2);
  bf16*  k_bf    = (bf16*) alloc((size_t)Mtok * D * 2);
  bf16*  v_bf    = (bf16*) alloc((size_t)Mtok * D * 2);
  bf16*  attn_bf = (bf16*) alloc((size_t)Mtok * D * 2);
  float* t1      = (float*)alloc((size_t)Mtok * D * 4);
  float* hf      = (float*)alloc((size_t)Mtok * D * 4);
  bf16*  h_bf    = (bf16*) alloc((size_t)Mtok * D * 2);
  bf16*  ff1_bf  = (bf16*) alloc((size_t)Mtok * F * 2);
  float* t2      = (float*)alloc((size_t)Mtok * D * 4);

  auto cvt = [&](const float* src, bf16* dst, size_t n) {
    cvt_f32_bf16<<<(unsigned)((n + 255) / 256), 256, 0, stream>>>(src, dst, (int)n);
  };
  cvt(x,    x_bf,    (size_t)Mtok * D);
  cvt(wqkv, wqkv_bf, (size_t)3 * D * D);
  cvt(wout, wout_bf, (size_t)D * D);
  cvt(w1,   w1_bf,   (size_t)F * D);
  cvt(w2,   w2_bf,   (size_t)D * F);

  // 1) QKV projection: [4096,1024] x [3072,1024]^T -> fp32 qkv
  gemm_bf16_nt<false><<<dim3(3 * D / 128, Mtok / 128), 256, 0, stream>>>(
      x_bf, wqkv_bf, bqkv, nullptr, qkv, nullptr, Mtok, 3 * D, D);

  // 2) RoPE + split to head-major bf16 q/k/v
  rope_split_kernel<<<(2 * 2048 * 16 * 32) / 256, 256, 0, stream>>>(
      qkv, q_bf, k_bf, v_bf);

  // 3) Flash attention -> token-major bf16 attn
  flash_attn_kernel<<<dim3(32, 16, 2), 128, 0, stream>>>(q_bf, k_bf, v_bf, attn_bf);

  // 4) Out-proj + residual(x) -> t1 fp32
  gemm_bf16_nt<false><<<dim3(D / 128, Mtok / 128), 256, 0, stream>>>(
      attn_bf, wout_bf, bout, x, t1, nullptr, Mtok, D, D);

  // 5) LN1 -> h (fp32 residual copy + bf16 GEMM input)
  layernorm_kernel<<<Mtok, 256, 0, stream>>>(t1, ln1g, ln1b, hf, h_bf);

  // 6) FFN up + ReLU -> bf16 only
  gemm_bf16_nt<true><<<dim3(F / 128, Mtok / 128), 256, 0, stream>>>(
      h_bf, w1_bf, b1, nullptr, nullptr, ff1_bf, Mtok, F, D);

  // 7) FFN down + residual(h) -> t2 fp32
  gemm_bf16_nt<false><<<dim3(D / 128, Mtok / 128), 256, 0, stream>>>(
      ff1_bf, w2_bf, b2, hf, t2, nullptr, Mtok, D, F);

  // 8) LN2 -> final fp32 output
  layernorm_kernel<<<Mtok, 256, 0, stream>>>(t2, ln2g, ln2b, out, nullptr);
}
